// slotGATConv_82188494176733
// MI455X (gfx1250) — matrix operand order repaired and checked by reading, not compiled, not run on hardware
//
#include <hip/hip_runtime.h>
#include <hip/hip_bf16.h>
#include <math.h>

#define NEG_SLOPE 0.2f
#define T_NT 3
#define NH   4
#define DIN  64
#define DOUT 32
#define EFW  64
#define NETY 5
#define NN   50000
#define NE   400000
#define FEAT_W (T_NT*DIN)          /* 192 */
#define OUT_W  (NH*T_NT*DOUT)      /* 384 */
#define HSLOT  (T_NT*DOUT)         /* 96  */

typedef __attribute__((ext_vector_type(16))) _Float16 v16h;
typedef __attribute__((ext_vector_type(8)))  _Float16 v8h;
typedef __attribute__((ext_vector_type(4)))  _Float16 v4h;
typedef __attribute__((ext_vector_type(8)))  float    v8f;
typedef __attribute__((ext_vector_type(4)))  unsigned int v4u;
typedef __attribute__((ext_vector_type(8)))  int      v8i;
typedef __attribute__((ext_vector_type(4)))  int      v4i;

// ---------------------------------------------------------------------------
// feat f32 -> f16 (one bandwidth pass); gives the TDM a tightly packed f16
// tensor [NN, 192] to DMA from, and halves A-tile bytes.
// ---------------------------------------------------------------------------
__global__ void feat_half_k(const float* __restrict__ feat,
                            _Float16* __restrict__ featH) {
    int tid = blockIdx.x * blockDim.x + threadIdx.x;
    if (tid >= NN * FEAT_W / 4) return;
    float4 v = ((const float4*)feat)[tid];
    v4h o = { (_Float16)v.x, (_Float16)v.y, (_Float16)v.z, (_Float16)v.w };
    ((v4h*)featH)[tid] = o;
}

// ---------------------------------------------------------------------------
// Pack fc / res_fc weights (f32 [3,64,128]) into f16 WMMA B-fragments.
// B 32x16 f16 layout used here (contiguous K, same order as A): lanes 0-15 hold K=0..15 (2 halves / VGPR),
// lanes 16-31 hold K=16..31; N = lane%16.
// ---------------------------------------------------------------------------
__global__ void pack_weights_k(const float* __restrict__ fc,
                               const float* __restrict__ res,
                               _Float16* __restrict__ bpack) {
    int tid = blockIdx.x * blockDim.x + threadIdx.x;
    if (tid >= 2 * T_NT * DIN * 128) return;
    int m    = tid / (T_NT * DIN * 128);
    int rem  = tid % (T_NT * DIN * 128);
    int t    = rem / (DIN * 128);
    int rem2 = rem % (DIN * 128);
    int k    = rem2 / 128;
    int col  = rem2 % 128;
    float v  = (m == 0 ? fc : res)[(t * DIN + k) * 128 + col];
    int kk   = k >> 5, krL = k & 31;
    int c    = col >> 4, n = col & 15;
    int lane = (krL >> 4) * 16 + n;
    int r    = (krL & 15) >> 1;
    int half = krL & 1;
    int frag = ((m * T_NT + t) * 2 + kk) * 8 + c;
    bpack[frag * 512 + lane * 16 + r * 2 + half] = (_Float16)v;
}

// ee_table[ty][h] = sum_j (edge_emb[ty] . fc_e_w[h*64+j]) * attn_e[h][j]
__global__ void ee_table_k(const float* __restrict__ emb,
                           const float* __restrict__ fcew,
                           const float* __restrict__ attn_e,
                           float* __restrict__ ee) {
    int tid = threadIdx.x;
    if (tid >= NETY * NH) return;
    int ty = tid / NH, h = tid % NH;
    float s = 0.f;
    for (int j = 0; j < EFW; ++j) {
        float acc = 0.f;
        for (int k = 0; k < EFW; ++k)
            acc += emb[ty * EFW + k] * fcew[(h * EFW + j) * EFW + k];
        s += acc * attn_e[h * EFW + j];
    }
    ee[ty * NH + h] = s;
}

__global__ void init_seg_k(float* __restrict__ emax, float* __restrict__ den) {
    int tid = blockIdx.x * blockDim.x + threadIdx.x;
    if (tid >= NN * NH) return;
    emax[tid] = -INFINITY;
    den[tid]  = 0.f;
}

// ---------------------------------------------------------------------------
// Node transform with TDM-staged A tiles.
// Wave 0 issues one tensor_load_to_lds (2-D D#: 16 rows x 192 f16, row
// stride 192 elems) pulling the node tile into LDS, waits on TENSORcnt,
// then the block barriers. 8 waves * 6 output 16x16 tiles each; K=64 as
// two v_wmma_f32_16x16x32_f16. ft -> workspace, residual -> d_out.
// ---------------------------------------------------------------------------
__global__ __launch_bounds__(256)
void node_gemm_k(const _Float16* __restrict__ featH,
                 const _Float16* __restrict__ bpack,
                 float* __restrict__ ftb,
                 float* __restrict__ outres) {
    __shared__ _Float16 a_lds[16 * FEAT_W];   // 6 KB f16 tile, [node][K]
    const int tid = threadIdx.x;
    const int nodeBase = blockIdx.x * 16;

    if (tid < 32) {
        // Build Tensor DMA Descriptor (CDNA5 ISA)
        unsigned long long ga =
            (unsigned long long)(uintptr_t)(featH + (size_t)nodeBase * FEAT_W);
        unsigned int lds_off = (unsigned int)(uintptr_t)(void*)a_lds; // LDS aperture low bits
        v4u g0 = { 1u,                                   // count=1, user mode
                   lds_off,                              // lds_addr (bytes)
                   (unsigned int)ga,                     // global_addr[31:0]
                   ((unsigned int)(ga >> 32) & 0x01FFFFFFu) | (2u << 30) }; // [56:32] | type=2
        v8i g1 = { (int)(1u << 16),                      // data_size=1 (2 bytes)
                   0,                                    // no atomic barrier
                   (int)0xC3500000u,                     // tensor_dim1[15:0]=50000 @ [95:80]
                   (int)(FEAT_W << 16),                  // tile_dim0=192 @ [127:112]
                   16,                                   // tile_dim1=16  @ [143:128]
                   FEAT_W,                               // tensor_dim0_stride=192 @ [191:160]
                   0, 0 };
        // tensor_dim0=192 lives at bits [79:48] -> low half in g1[1] upper 16
        g1[1] = (int)((unsigned int)FEAT_W << 16);
        v4i z4 = { 0, 0, 0, 0 };
#if defined(__clang_major__) && (__clang_major__ >= 23)
        v8i z8 = { 0, 0, 0, 0, 0, 0, 0, 0 };
        __builtin_amdgcn_tensor_load_to_lds(g0, g1, z4, z4, z8, 0);
#else
        __builtin_amdgcn_tensor_load_to_lds(g0, g1, z4, z4, 0);
#endif
        __builtin_amdgcn_s_wait_tensorcnt(0);
    }
    __syncthreads();

    const int lane = tid & 31;
    const int na   = lane & 15;            // A row (M)
    const int kgrp = (lane >> 4) * 16;     // A K-group base (contiguous K = 16h + s)

    for (int it = 0; it < 6; ++it) {
        const int tile = (tid >> 5) * 6 + it;
        const int m    = tile / 24;        // 0: fc path, 1: residual path
        const int rem  = tile % 24;
        const int t    = rem >> 3;         // slot
        const int c    = rem & 7;          // 16-wide column tile
        v8f acc = {};
#pragma unroll
        for (int kk = 0; kk < 2; ++kk) {
            const _Float16* ap = &a_lds[na * FEAT_W + t * DIN + kk * 32 + kgrp];
            v8h lo = *(const v8h*)ap;              // ds_load_b128
            v8h hi = *(const v8h*)(ap + 8);        // ds_load_b128
            v16h a;
#pragma unroll
            for (int q = 0; q < 8; ++q) { a[q] = lo[q]; a[q + 8] = hi[q]; }
            const int frag = ((m * T_NT + t) * 2 + kk) * 8 + c;
            v16h b = *(const v16h*)(bpack + frag * 512 + lane * 16);
            acc = __builtin_amdgcn_wmma_f32_16x16x32_f16(
                      false, a, false, b, (short)0, acc, false, false);
        }
        // C layout: VGPR r -> row M = r / r+8, N = lane%16. Rows are a
        // constant 384-float stride apart -> single base + immediate offsets.
        float* dstb   = (m == 0) ? ftb : outres;
        const int h   = c >> 1;
        const int db  = (c & 1) * 16 + (lane & 15);
        const int ro  = (lane < 16) ? 0 : 8;
        float* pb = dstb + (size_t)(nodeBase + ro) * OUT_W + h * HSLOT + t * DOUT + db;
#pragma unroll
        for (int r = 0; r < 8; ++r)
            pb[r * OUT_W] = acc[r];
    }
}

// el/er per (node, head): dot(ft[n,h,:96], attn_l/attn_r[h,:96])
__global__ void node_attn_k(const float* __restrict__ ftb,
                            const float* __restrict__ attn_l,
                            const float* __restrict__ attn_r,
                            float* __restrict__ el, float* __restrict__ er) {
    int tid = blockIdx.x * blockDim.x + threadIdx.x;
    if (tid >= NN * NH) return;
    int h = tid & 3;
    const float4* f  = (const float4*)(ftb + (size_t)tid * HSLOT);
    const float4* al = (const float4*)(attn_l + h * HSLOT);
    const float4* ar = (const float4*)(attn_r + h * HSLOT);
    float sl = 0.f, sr = 0.f;
#pragma unroll
    for (int j = 0; j < HSLOT / 4; ++j) {
        float4 v = f[j], a = al[j], b = ar[j];
        sl += v.x * a.x + v.y * a.y + v.z * a.z + v.w * a.w;
        sr += v.x * b.x + v.y * b.y + v.z * b.z + v.w * b.w;
    }
    el[tid] = sl; er[tid] = sr;
}

// logits + leaky relu + segment max (f32 atomic-max via int bit trick)
__global__ void edge_logit_k(const int* __restrict__ etype,
                             const int* __restrict__ src, const int* __restrict__ dst,
                             const float* __restrict__ el, const float* __restrict__ er,
                             const float* __restrict__ ee,
                             float* __restrict__ ebuf, float* __restrict__ emax) {
    int tid = blockIdx.x * blockDim.x + threadIdx.x;
    if (tid >= NE * NH) return;
    int e = tid >> 2, h = tid & 3;
    float v = el[src[e] * NH + h] + er[dst[e] * NH + h] + ee[etype[e] * NH + h];
    v = (v >= 0.f) ? v : NEG_SLOPE * v;
    ebuf[tid] = v;
    float* addr = &emax[dst[e] * NH + h];
    if (v >= 0.f) atomicMax((int*)addr, __float_as_int(v));
    else          atomicMin((unsigned int*)addr, __float_as_uint(v));
}

__global__ void edge_exp_k(const int* __restrict__ dst,
                           const float* __restrict__ emax,
                           float* __restrict__ ebuf, float* __restrict__ den) {
    int tid = blockIdx.x * blockDim.x + threadIdx.x;
    if (tid >= NE * NH) return;
    int e = tid >> 2, h = tid & 3;
    float ex = __expf(ebuf[tid] - emax[dst[e] * NH + h]);
    ebuf[tid] = ex;
    atomicAdd(&den[dst[e] * NH + h], ex);
}

__global__ void edge_coef_k(const int* __restrict__ dst,
                            const float* __restrict__ den, float* __restrict__ ebuf) {
    int tid = blockIdx.x * blockDim.x + threadIdx.x;
    if (tid >= NE * NH) return;
    int e = tid >> 2, h = tid & 3;
    ebuf[tid] = ebuf[tid] / den[dst[e] * NH + h];
}

// one block per edge (384 threads): gather ft[src]*a, scatter-add into out[dst]
__global__ __launch_bounds__(OUT_W)
void edge_scatter_k(const int* __restrict__ src, const int* __restrict__ dst,
                    const float* __restrict__ ebuf,
                    const float* __restrict__ ftb, float* __restrict__ out) {
    const int e   = blockIdx.x;
    const int idx = threadIdx.x;
    const int h   = idx / HSLOT;
    float a = ebuf[e * NH + h];
    atomicAdd(&out[(size_t)dst[e] * OUT_W + idx],
              a * ftb[(size_t)src[e] * OUT_W + idx]);
}

// ---------------------------------------------------------------------------
extern "C" void kernel_launch(void* const* d_in, const int* in_sizes, int n_in,
                              void* d_out, int out_size, void* d_ws, size_t ws_size,
                              hipStream_t stream) {
    const float* feat    = (const float*)d_in[0];
    const int*   e_feat  = (const int*)  d_in[1];
    const int*   src     = (const int*)  d_in[2];
    const int*   dst     = (const int*)  d_in[3];
    const float* fc      = (const float*)d_in[4];
    const float* res_fc  = (const float*)d_in[5];
    const float* emb     = (const float*)d_in[6];
    const float* fc_e_w  = (const float*)d_in[7];
    const float* attn_l  = (const float*)d_in[8];
    const float* attn_r  = (const float*)d_in[9];
    const float* attn_e  = (const float*)d_in[10];
    float* out = (float*)d_out;

    uint8_t* p = (uint8_t*)d_ws;
    auto carve = [&](size_t bytes) {
        uint8_t* r = p;
        p += (bytes + 255) & ~(size_t)255;
        return (void*)r;
    };
    float*    ftb   = (float*)   carve((size_t)NN * OUT_W * 4);   // 76.8 MB
    _Float16* featH = (_Float16*)carve((size_t)NN * FEAT_W * 2);  // 19.2 MB
    float*    ebuf  = (float*)   carve((size_t)NE * NH * 4);      // 6.4 MB
    float*    el    = (float*)   carve((size_t)NN * NH * 4);
    float*    er    = (float*)   carve((size_t)NN * NH * 4);
    float*    emax  = (float*)   carve((size_t)NN * NH * 4);
    float*    den   = (float*)   carve((size_t)NN * NH * 4);
    _Float16* bpack = (_Float16*)carve((size_t)96 * 512 * 2);     // 96 KB
    float*    eetab = (float*)   carve((size_t)NETY * NH * 4);
    (void)ws_size; (void)n_in; (void)in_sizes; (void)out_size;

    const int B = 256;
    feat_half_k<<<(NN * FEAT_W / 4 + B - 1) / B, B, 0, stream>>>(feat, featH);
    pack_weights_k<<<(2 * T_NT * DIN * 128 + B - 1) / B, B, 0, stream>>>(fc, res_fc, bpack);
    ee_table_k<<<1, 32, 0, stream>>>(emb, fc_e_w, attn_e, eetab);
    init_seg_k<<<(NN * NH + B - 1) / B, B, 0, stream>>>(emax, den);

    node_gemm_k<<<NN / 16, 256, 0, stream>>>(featH, bpack, ftb, out);

    node_attn_k<<<(NN * NH + B - 1) / B, B, 0, stream>>>(ftb, attn_l, attn_r, el, er);
    edge_logit_k<<<(NE * NH + B - 1) / B, B, 0, stream>>>(e_feat, src, dst, el, er,
                                                          eetab, ebuf, emax);
    edge_exp_k<<<(NE * NH + B - 1) / B, B, 0, stream>>>(dst, emax, ebuf, den);
    edge_coef_k<<<(NE * NH + B - 1) / B, B, 0, stream>>>(dst, den, ebuf);
    edge_scatter_k<<<NE, OUT_W, 0, stream>>>(src, dst, ebuf, ftb, out);
}
